// SwinTransformerWithAttn_1932735283793
// MI455X (gfx1250) — matrix operand order repaired and checked by reading, not compiled, not run on hardware
//
#include <hip/hip_runtime.h>
#include <hip/hip_bf16.h>
#include <math.h>

// ---------------- CDNA5 WMMA plumbing ----------------
typedef __attribute__((ext_vector_type(16))) __bf16 v16bf;
typedef __attribute__((ext_vector_type(8)))  __bf16 v8bf;
typedef __attribute__((ext_vector_type(8)))  float  v8f;
typedef __attribute__((ext_vector_type(4)))  float  v4f;
typedef __attribute__((ext_vector_type(4)))  unsigned int v4u;

__device__ __forceinline__ unsigned short bf16bits(float f) {
    unsigned u = __float_as_uint(f);
    u += 0x7FFFu + ((u >> 16) & 1u);           // round-to-nearest-even
    return (unsigned short)(u >> 16);
}
__device__ __forceinline__ __bf16 f2bf(float f) {
    unsigned short s = bf16bits(f);
    return __builtin_bit_cast(__bf16, s);
}
#if __has_builtin(__builtin_amdgcn_cvt_pk_bf16_f32)
typedef __attribute__((ext_vector_type(2))) __bf16 v2bf;
__device__ __forceinline__ unsigned pack2(float lo, float hi) {
    v2bf r = __builtin_amdgcn_cvt_pk_bf16_f32(lo, hi);
    return __builtin_bit_cast(unsigned, r);
}
#else
__device__ __forceinline__ unsigned pack2(float lo, float hi) {
    return (unsigned)bf16bits(lo) | ((unsigned)bf16bits(hi) << 16);
}
#endif
__device__ __forceinline__ v4u pack8(v4f a, v4f b) {
    v4u u;
    u[0] = pack2(a[0], a[1]); u[1] = pack2(a[2], a[3]);
    u[2] = pack2(b[0], b[1]); u[3] = pack2(b[2], b[3]);
    return u;
}

__device__ __forceinline__ v8f wmma_bf16(v16bf a, v16bf b, v8f c) {
    return __builtin_amdgcn_wmma_f32_16x16x32_bf16(
        false, a, false, b, (short)0, c, false, false);
}

// ---- async global -> LDS copy (16 bytes), ASYNCcnt-tracked ----
__device__ __forceinline__ void async_copy16(const __bf16* gsrc, __bf16* ldst) {
    unsigned loff = (unsigned)(unsigned long long)ldst;   // low 32 bits = LDS offset
    asm volatile("global_load_async_to_lds_b128 %0, %1, off"
                 :: "v"(loff), "v"(gsrc) : "memory");
}
__device__ __forceinline__ void wait_async() {
    asm volatile("s_wait_asynccnt 0" ::: "memory");
}

// A fragment: rows m0..m0+15, K-window of 32 at `base` (row stride `stride` bf16).
__device__ __forceinline__ v16bf load_a_frag(const __bf16* base, int stride, int m0, int lane) {
    int r  = m0 + (lane & 15);
    int kh = (lane & 16) ? 8 : 0;
    const __bf16* p = base + r * stride + kh;
    union { v16bf v; v8bf h[2]; } u;
    u.h[0] = *(const v8bf*)(p);
    u.h[1] = *(const v8bf*)(p + 16);
    return u.v;
}
// B fragment: cols n0..n0+15; `base` stores B^T (row n, contiguous K).
__device__ __forceinline__ v16bf load_b_frag(const __bf16* base, int stride, int n0, int lane) {
    int c  = n0 + (lane & 15);
    int ko = (lane & 16) ? 8 : 0;
    const __bf16* p = base + c * stride + ko;
    union { v16bf v; v8bf h[2]; } u;
    u.h[0] = *(const v8bf*)(p);
    u.h[1] = *(const v8bf*)(p + 16);
    return u.v;
}

// ---------------- Kernel 0: fp32 -> bf16 convert (8 elems / thread) ----------------
__global__ void cvt_kernel(const float* __restrict__ in, __bf16* __restrict__ out, long long n8) {
    long long i = (long long)blockIdx.x * 256 + threadIdx.x;
    if (i >= n8) return;
    const float* p = in + i * 8;
    v4f f0 = *(const v4f*)p, f1 = *(const v4f*)(p + 4);
    *(v4u*)(out + i * 8) = pack8(f0, f1);
}

// ---------------- Kernel 1: patch embed (conv4x4 s4) ----------------
__global__ void patch_embed_kernel(const float* __restrict__ x,
                                   const float* __restrict__ w,
                                   const float* __restrict__ bias,
                                   float* __restrict__ out) {
    const long long total = 32LL * 3136 * 96;
    long long idx = (long long)blockIdx.x * 256 + threadIdx.x;
    if (idx >= total) return;
    int c = (int)(idx % 96);
    long long t = idx / 96;
    int l = (int)(t % 3136);
    int b = (int)(t / 3136);
    int py = l / 56, px = l % 56;
    float s = bias[c];
    for (int ic = 0; ic < 4; ++ic)
        for (int ky = 0; ky < 4; ++ky) {
            const float* xr = x + (((long long)b * 4 + ic) * 224 + (py * 4 + ky)) * 224 + px * 4;
            const float* wr = w + ((c * 4 + ic) * 4 + ky) * 4;
            v4f xv = *(const v4f*)xr;
            s += xv[0] * wr[0] + xv[1] * wr[1] + xv[2] * wr[2] + xv[3] * wr[3];
        }
    out[idx] = s;
}

// ---------------- Kernel 2: LayerNorm (one wave per row), fp32 or bf16 out ----------------
__global__ void layernorm_kernel(const float* __restrict__ in,
                                 const float* __restrict__ w,
                                 const float* __restrict__ b,
                                 float* __restrict__ outf, __bf16* __restrict__ outb,
                                 int M, int C) {
    int wave = (int)((blockIdx.x * (long long)blockDim.x + threadIdx.x) >> 5);
    int lane = threadIdx.x & 31;
    if (wave >= M) return;
    const float* row = in + (long long)wave * C;
    float s = 0.f, s2 = 0.f;
    for (int c = lane; c < C; c += 32) { float v = row[c]; s += v; s2 += v * v; }
    for (int o = 16; o; o >>= 1) { s += __shfl_xor(s, o, 32); s2 += __shfl_xor(s2, o, 32); }
    float mu  = s / (float)C;
    float var = s2 / (float)C - mu * mu;
    float r   = rsqrtf(var + 1e-5f);
    for (int c = lane; c < C; c += 32) {
        float y = (row[c] - mu) * r * w[c] + b[c];
        if (outf) outf[(long long)wave * C + c] = y;
        else      outb[(long long)wave * C + c] = f2bf(y);
    }
}

// ---------------- Kernel 3: window partition (+ flattened L/C roll), bf16 out ----------------
__global__ void winpart_kernel(const float* __restrict__ in, __bf16* __restrict__ out,
                               int B, int H, int C, int shift) {
    int L = H * H, wps = H / 7, nW = wps * wps;
    long long total = (long long)B * L * C;
    long long idx = (long long)blockIdx.x * 256 + threadIdx.x;
    if (idx >= total) return;
    int c = (int)(idx % C);
    long long t = idx / C;
    int n = (int)(t % 49);
    long long wi = t / 49;
    int w2 = (int)(wi % nW);
    int b  = (int)(wi / nW);
    int wy = w2 / wps, wx = w2 % wps;
    int rr = wy * 7 + n / 7, cc = wx * 7 + n % 7;
    int l = rr * H + cc;
    int ls = l, cs = c;
    if (shift) { ls = (l + shift) % L; cs = (c + shift) % C; }
    out[idx] = f2bf(in[((long long)b * L + ls) * C + cs]);
}

// ---------------- Kernel 4: WMMA GEMM (bf16 in), async-LDS staged, double-buffered ----------------
// 128x64 block tile, 8 waves * 4 WMMA tiles. REQUIRES K % 32 == 0.
__global__ void __launch_bounds__(256)
gemm_kernel(const __bf16* __restrict__ X, const __bf16* __restrict__ W,
            const float* __restrict__ bias, const float* __restrict__ resid,
            float* __restrict__ outf, __bf16* __restrict__ outb,
            int M, int N, int K, int act) {
    __shared__ __align__(16) __bf16 At[2][128][32];   // 16 KB
    __shared__ __align__(16) __bf16 Bt[2][64][32];    //  8 KB
    int tid = threadIdx.x, lane = tid & 31, w = tid >> 5;
    int wm = w & 3, wn = w >> 2;
    int mblk = blockIdx.y * 128, nblk = blockIdx.x * 64;
    v8f acc00 = {}, acc01 = {}, acc10 = {}, acc11 = {};
    int rr_[3], ck_[3], isA_[3];
    #pragma unroll
    for (int i = 0; i < 3; ++i) {
        int chunk = tid + i * 256;                    // 768 chunks: 128*4 A + 64*4 B
        rr_[i] = chunk >> 2;
        ck_[i] = (chunk & 3) << 3;
        isA_[i] = (rr_[i] < 128);
    }
    // pre-zero both buffers so OOB tail rows read as 0 (copies are skipped for them)
    {
        v4u z = {};
        #pragma unroll
        for (int i = 0; i < 4; ++i) ((v4u*)At)[tid + i * 256] = z;
        #pragma unroll
        for (int i = 0; i < 2; ++i) ((v4u*)Bt)[tid + i * 256] = z;
    }
    __syncthreads();
    auto issue = [&](int kk, int buf) {
        #pragma unroll
        for (int i = 0; i < 3; ++i) {
            int gr  = isA_[i] ? (mblk + rr_[i]) : (nblk + (rr_[i] - 128));
            int lim = isA_[i] ? M : N;
            if (gr < lim) {
                const __bf16* src = (isA_[i] ? X : W) + (long long)gr * K + kk + ck_[i];
                __bf16* dst = isA_[i] ? &At[buf][rr_[i]][ck_[i]]
                                      : &Bt[buf][rr_[i] - 128][ck_[i]];
                async_copy16(src, dst);
            }
        }
    };
    issue(0, 0);
    wait_async();
    __syncthreads();
    int cur = 0;
    for (int kk = 0; kk < K; kk += 32) {
        bool more = (kk + 32) < K;
        if (more) issue(kk + 32, cur ^ 1);            // DMA fills next buffer during WMMAs
        v16bf a0 = load_a_frag(&At[cur][0][0], 32, wm * 16,      lane);
        v16bf a1 = load_a_frag(&At[cur][0][0], 32, wm * 16 + 64, lane);
        v16bf b0 = load_b_frag(&Bt[cur][0][0], 32, wn * 32,      lane);
        v16bf b1 = load_b_frag(&Bt[cur][0][0], 32, wn * 32 + 16, lane);
        acc00 = wmma_bf16(a0, b0, acc00);
        acc01 = wmma_bf16(a0, b1, acc01);
        acc10 = wmma_bf16(a1, b0, acc10);
        acc11 = wmma_bf16(a1, b1, acc11);
        if (more) {
            wait_async();
            __syncthreads();
            cur ^= 1;
        }
    }
    int col0 = nblk + wn * 32 + (lane & 15);
    int r0   = mblk + wm * 16 + ((lane & 16) ? 8 : 0);
    #pragma unroll
    for (int mi = 0; mi < 2; ++mi) {
        #pragma unroll
        for (int e = 0; e < 8; ++e) {
            int r = r0 + mi * 64 + e;
            if (r >= M) continue;
            #pragma unroll
            for (int ni = 0; ni < 2; ++ni) {
                int col = col0 + ni * 16;
                if (col >= N) continue;
                float y = mi ? (ni ? acc11[e] : acc10[e]) : (ni ? acc01[e] : acc00[e]);
                if (bias) y += bias[col];
                if (act == 1) y = 0.5f * y * (1.f + erff(y * 0.70710678118654752f));
                long long o = (long long)r * N + col;
                if (outf) {
                    if (resid) y += resid[o];
                    outf[o] = y;
                } else {
                    outb[o] = f2bf(y);
                }
            }
        }
    }
}

// ---------------- Kernel 5: fused window attention per (window, head) ----------------
__global__ void __launch_bounds__(256)
attn_kernel(const __bf16* __restrict__ qkv, const float* __restrict__ rpb,
            const int* __restrict__ relidx, const float* __restrict__ mask,
            __bf16* __restrict__ O, float* __restrict__ attn_out,
            int nh, int C, int nW) {
    __shared__ __align__(16) __bf16 Qb[64][32];
    __shared__ __align__(16) __bf16 Kb[64][32];
    __shared__ __align__(16) __bf16 Vt[32][64];
    __shared__ __align__(16) float  Sb[64][64];
    __shared__ __align__(16) __bf16 Pb[64][64];
    int win = blockIdx.x, head = blockIdx.y;
    int tid = threadIdx.x, lane = tid & 31, wv = tid >> 5;
    int C3 = 3 * C;
    const float scale = 0.17677669529663687f;     // 32^-0.5, folded into S epilogue
    {
        int r = tid >> 2, ck = (tid & 3) << 3;
        v4u q = {}, k = {}, v = {};
        if (r < 49) {
            const __bf16* bq = qkv + ((long long)win * 49 + r) * C3 + head * 32 + ck;
            q = *(const v4u*)(bq);
            k = *(const v4u*)(bq + C);
            v = *(const v4u*)(bq + 2 * C);
        }
        *(v4u*)&Qb[r][ck] = q;
        *(v4u*)&Kb[r][ck] = k;
        union { v4u u; __bf16 e[8]; } vu; vu.u = v;
        #pragma unroll
        for (int e = 0; e < 8; ++e) Vt[ck + e][r] = vu.e[e];
    }
    __syncthreads();
    const float* mrow = mask ? (mask + (long long)(win % nW) * 49 * 49) : (const float*)0;
    #pragma unroll
    for (int tt = 0; tt < 2; ++tt) {
        int t = wv * 2 + tt;
        int m0 = (t >> 2) * 16, n0 = (t & 3) * 16;
        v16bf a  = load_a_frag(&Qb[0][0], 32, m0, lane);
        v16bf bb = load_b_frag(&Kb[0][0], 32, n0, lane);
        v8f acc = {};
        acc = wmma_bf16(a, bb, acc);
        int col = n0 + (lane & 15);
        int rb  = m0 + ((lane & 16) ? 8 : 0);
        #pragma unroll
        for (int e = 0; e < 8; ++e) {
            int rr = rb + e;
            float s;
            if (rr < 49 && col < 49) {
                s = acc[e] * scale + rpb[(long long)relidx[rr * 49 + col] * nh + head];
                if (mrow) s += mrow[rr * 49 + col];
            } else s = -1e30f;
            Sb[rr][col] = s;
        }
    }
    __syncthreads();
    for (int r = wv; r < 49; r += 8) {
        float v0 = Sb[r][lane], v1 = Sb[r][lane + 32];
        float mx = fmaxf(v0, v1);
        for (int o = 16; o; o >>= 1) mx = fmaxf(mx, __shfl_xor(mx, o, 32));
        float e0 = __expf(v0 - mx), e1 = __expf(v1 - mx);
        float sm = e0 + e1;
        for (int o = 16; o; o >>= 1) sm += __shfl_xor(sm, o, 32);
        float inv = 1.f / sm;
        float p0 = e0 * inv, p1 = e1 * inv;
        Pb[r][lane]      = f2bf(p0);
        Pb[r][lane + 32] = f2bf(p1);
        long long ab = (((long long)win * nh + head) * 49 + r) * 49;
        if (lane < 49)      attn_out[ab + lane]      = p0;
        if (lane + 32 < 49) attn_out[ab + lane + 32] = p1;
    }
    __syncthreads();
    {
        int m0 = (wv >> 1) * 16, nf = (wv & 1) * 16;
        v8f acc = {};
        #pragma unroll
        for (int ks = 0; ks < 64; ks += 32) {
            v16bf a  = load_a_frag(&Pb[0][ks], 64, m0, lane);
            v16bf bb = load_b_frag(&Vt[0][ks], 64, nf, lane);
            acc = wmma_bf16(a, bb, acc);
        }
        int col = nf + (lane & 15);
        int rb  = m0 + ((lane & 16) ? 8 : 0);
        #pragma unroll
        for (int e = 0; e < 8; ++e) {
            int rr = rb + e;
            if (rr < 49)
                O[((long long)win * 49 + rr) * C + head * 32 + col] = f2bf(acc[e]);
        }
    }
}

// ---------------- Kernel 6: window merge + roll-back + residual add (in place) ----------------
__global__ void merge_back_add_kernel(float* __restrict__ x, const float* __restrict__ P,
                                      int B, int H, int C, int shift) {
    int L = H * H, wps = H / 7;
    long long total = (long long)B * L * C;
    long long idx = (long long)blockIdx.x * 256 + threadIdx.x;
    if (idx >= total) return;
    int c = (int)(idx % C);
    long long t = idx / C;
    int l = (int)(t % L);
    int b = (int)(t / L);
    int lm = l, cm = c;
    if (shift) { lm = (l - shift + L) % L; cm = (c - shift + C) % C; }
    int rr = lm / H, cc = lm % H;
    int wy = rr / 7, py = rr % 7, wx = cc / 7, px = cc % 7;
    long long win = ((long long)b * wps + wy) * wps + wx;
    x[idx] += P[(win * 49 + py * 7 + px) * C + cm];
}

// ---------------- Kernel 7: patch-merge gather ----------------
__global__ void merge_gather_kernel(const float* __restrict__ x, float* __restrict__ out,
                                    int B, int H, int C) {
    int H2 = H / 2;
    long long total = (long long)B * H2 * H2 * 4 * C;
    long long idx = (long long)blockIdx.x * 256 + threadIdx.x;
    if (idx >= total) return;
    int c4 = (int)(idx % (4 * C));
    long long t = idx / (4 * C);
    int p = (int)(t % (H2 * H2));
    int b = (int)(t / (H2 * H2));
    int yi = p / H2, xi = p % H2;
    int g = c4 / C, c = c4 % C;
    int dy = g & 1, dx = g >> 1;
    out[idx] = x[((long long)b * H * H + (2 * yi + dy) * H + (2 * xi + dx)) * C + c];
}

// ---------------- Host orchestration ----------------
static inline int ceil_div(long long a, long long b) { return (int)((a + b - 1) / b); }

extern "C" void kernel_launch(void* const* d_in, const int* in_sizes, int n_in,
                              void* d_out, int out_size, void* d_ws, size_t ws_size,
                              hipStream_t stream) {
    (void)in_sizes; (void)n_in; (void)out_size; (void)ws_size;
    const int DEPTH[4] = {2, 2, 6, 2};
    const int NHEAD[4] = {3, 6, 12, 24};
    const int Bsz = 32;

    const float* x    = (const float*)d_in[0];
    const float* pe_w = (const float*)d_in[1];
    const float* pe_b = (const float*)d_in[2];
    const float* nw   = (const float*)d_in[3];
    const float* nb   = (const float*)d_in[4];
    int blk_base[4][6];
    int mrg_base[3];
    int base = 5;
    for (int i = 0; i < 4; ++i) {
        for (int j = 0; j < DEPTH[i]; ++j) { blk_base[i][j] = base; base += 13; }
        if (i < 3) { mrg_base[i] = base; base += 3; }
    }
    const int* relidx = (const int*)d_in[base];
    const float* masks[3] = { (const float*)d_in[base + 1],
                              (const float*)d_in[base + 2],
                              (const float*)d_in[base + 3] };

    const long long SLOT = 9633792LL;              // 32*3136*96
    float* A  = (float*)d_ws;                      // fp32 residual stream
    float* Bb = A  + SLOT;                         // LN out (fp32 or bf16 alias)
    float* Cc = Bb + SLOT;                         // windows bf16 / proj fp32
    float* E  = Cc + SLOT;                         // attention O (bf16 alias)
    float* D  = E  + SLOT;                         // qkv / MLP hidden (bf16 alias, 4x slot)
    __bf16* WBUF = (__bf16*)(D + 38535168LL);      // weight bf16 scratch (reused, stream-ordered)

    auto cvtw = [&](const float* src, long long nelem) -> const __bf16* {
        long long n8 = nelem / 8;
        cvt_kernel<<<ceil_div(n8, 256), 256, 0, stream>>>(src, WBUF, n8);
        return (const __bf16*)WBUF;
    };

    float* out = (float*)d_out;
    long long attn_off = 32LL * 49 * 768;

    {
        long long total = 32LL * 3136 * 96;
        patch_embed_kernel<<<ceil_div(total, 256), 256, 0, stream>>>(x, pe_w, pe_b, A);
    }

    int res = 56;
    for (int i = 0; i < 4; ++i) {
        int C  = 96 << i;
        int nh = NHEAD[i];
        int L  = res * res;
        int M  = Bsz * L;
        int wps = res / 7;
        int nW  = wps * wps;
        int Bwin = Bsz * nW;
        long long elems = (long long)M * C;

        for (int j = 0; j < DEPTH[i]; ++j) {
            const int bp = blk_base[i][j];
            const float* n1w  = (const float*)d_in[bp + 0];
            const float* n1b  = (const float*)d_in[bp + 1];
            const float* qkvw = (const float*)d_in[bp + 2];
            const float* qkvb = (const float*)d_in[bp + 3];
            const float* rpb  = (const float*)d_in[bp + 4];
            const float* pjw  = (const float*)d_in[bp + 5];
            const float* pjb  = (const float*)d_in[bp + 6];
            const float* n2w  = (const float*)d_in[bp + 7];
            const float* n2b  = (const float*)d_in[bp + 8];
            const float* f1w  = (const float*)d_in[bp + 9];
            const float* f1b  = (const float*)d_in[bp + 10];
            const float* f2w  = (const float*)d_in[bp + 11];
            const float* f2b  = (const float*)d_in[bp + 12];

            int shift = (j % 2 == 1 && res > 7) ? 3 : 0;
            const float* mptr = shift ? masks[i] : (const float*)0;

            layernorm_kernel<<<ceil_div(M, 8), 256, 0, stream>>>(
                A, n1w, n1b, Bb, (__bf16*)0, M, C);
            winpart_kernel<<<ceil_div(elems, 256), 256, 0, stream>>>(
                Bb, (__bf16*)Cc, Bsz, res, C, shift);
            gemm_kernel<<<dim3(ceil_div(3 * C, 64), ceil_div(M, 128)), 256, 0, stream>>>(
                (const __bf16*)Cc, cvtw(qkvw, 3LL * C * C), qkvb, (const float*)0,
                (float*)0, (__bf16*)D, M, 3 * C, C, 0);
            attn_kernel<<<dim3(Bwin, nh), 256, 0, stream>>>(
                (const __bf16*)D, rpb, relidx, mptr, (__bf16*)E, out + attn_off, nh, C, nW);
            attn_off += (long long)Bwin * nh * 49 * 49;
            gemm_kernel<<<dim3(ceil_div(C, 64), ceil_div(M, 128)), 256, 0, stream>>>(
                (const __bf16*)E, cvtw(pjw, (long long)C * C), pjb, (const float*)0,
                Cc, (__bf16*)0, M, C, C, 0);
            merge_back_add_kernel<<<ceil_div(elems, 256), 256, 0, stream>>>(
                A, Cc, Bsz, res, C, shift);
            layernorm_kernel<<<ceil_div(M, 8), 256, 0, stream>>>(
                A, n2w, n2b, (float*)0, (__bf16*)Bb, M, C);
            gemm_kernel<<<dim3(ceil_div(4 * C, 64), ceil_div(M, 128)), 256, 0, stream>>>(
                (const __bf16*)Bb, cvtw(f1w, 4LL * C * C), f1b, (const float*)0,
                (float*)0, (__bf16*)D, M, 4 * C, C, 1 /*gelu*/);
            gemm_kernel<<<dim3(ceil_div(C, 64), ceil_div(M, 128)), 256, 0, stream>>>(
                (const __bf16*)D, cvtw(f2w, 4LL * C * C), f2b, A /*residual*/,
                A, (__bf16*)0, M, C, 4 * C, 0);
        }

        if (i < 3) {
            const int mp = mrg_base[i];
            const float* mnw = (const float*)d_in[mp + 0];
            const float* mnb = (const float*)d_in[mp + 1];
            const float* rdw = (const float*)d_in[mp + 2];
            int M2 = M / 4;
            merge_gather_kernel<<<ceil_div(elems, 256), 256, 0, stream>>>(A, Cc, Bsz, res, C);
            layernorm_kernel<<<ceil_div(M2, 8), 256, 0, stream>>>(
                Cc, mnw, mnb, (float*)0, (__bf16*)Bb, M2, 4 * C);
            gemm_kernel<<<dim3(ceil_div(2 * C, 64), ceil_div(M2, 128)), 256, 0, stream>>>(
                (const __bf16*)Bb, cvtw(rdw, 8LL * C * C), (const float*)0, (const float*)0,
                A, (__bf16*)0, M2, 2 * C, 4 * C, 0);
            res /= 2;
        }
    }

    layernorm_kernel<<<ceil_div(32 * 49, 8), 256, 0, stream>>>(
        A, nw, nb, out, (__bf16*)0, 32 * 49, 768);
}
